// ECA_69544110457539
// MI455X (gfx1250) — matrix operand order repaired and checked by reading, not compiled, not run on hardware
//
#include <hip/hip_runtime.h>

// ---- problem constants (from reference) ----
#define XS_     128
#define HST     160     // HS + T
#define NST     160     // NS + T
#define EE      512     // E
#define C3_     1536    // 3*E
#define ROWS    16      // rows (b,t pairs) per workgroup
#define THREADS 256     // 8 wave32
#define KTOT    448     // 128 + 160 + 160

// B-fragment workspace: 448 fragments x 512 bf16 (1 KB) each
#define FB0     0       // gemm0: 32 ntiles * 4 ksteps = 128 frags
#define FB1     128     // gemm1: 32 * 5 = 160 frags
#define FB2     288     // gemm2: 32 * 5 = 160 frags
#define NFRAG_B 448
#define WS_BYTES (NFRAG_B * 512 * 2)   // 458752

// A-fragment LDS: 14 fragments x 512 bf16
#define FA0     0
#define FA1     4
#define FA2     9

typedef __attribute__((ext_vector_type(16))) __bf16 v16bf;
typedef __attribute__((ext_vector_type(8)))  float  v8f;

// LDS layout (bytes), dynamic shared memory
#define OFF_A    0            // ushort[14*512]      = 14336 B (fragment-swizzled)
#define OFF_PERM 14336        // int[1536]           =  6144 B
#define OFF_INV  20480        // int[1536]           =  6144 B
#define OFF_XHN  26624        // float[16*1536]      = 98304 B
#define OFF_S    124928       // float[16*512]       = 32768 B
#define SMEM_BYTES 157696

__device__ __forceinline__ unsigned short f2bfu(float f) {
    union { float f; unsigned u; } c; c.f = f;
    unsigned u = c.u;
    u += 0x7FFFu + ((u >> 16) & 1u);   // round-to-nearest-even
    return (unsigned short)(u >> 16);
}

// ---------------------------------------------------------------------------
// Kernel 1: repack Wx/Wh/Wn (f32, [K][512]) into bf16 B-fragments in d_ws.
// B-fragment layout (32x16 bf16): lane = n%16 + 16*((kr/8)%2), elem = 8*(kr/16) + kr%8.
// Fragment index: FBg + (n/16)*KSTEPS + (k/32). Lane data contiguous 32 B.
// ---------------------------------------------------------------------------
extern "C" __global__ __launch_bounds__(THREADS)
void repack_w_kernel(const float* __restrict__ Wx, const float* __restrict__ Wh,
                     const float* __restrict__ Wn, unsigned short* __restrict__ wsb)
{
    int e = blockIdx.x * THREADS + threadIdx.x;      // 0 .. 448*512-1
    if (e >= KTOT * EE) return;
    int kt = e >> 9;            // packed K index 0..447
    int n  = e & 511;
    float val; int kl, fb, ksteps;
    if (kt < 128)      { kl = kt;       val = Wx[(size_t)kl * EE + n]; fb = FB0; ksteps = 4; }
    else if (kt < 288) { kl = kt - 128; val = Wh[(size_t)kl * EE + n]; fb = FB1; ksteps = 5; }
    else               { kl = kt - 288; val = Wn[(size_t)kl * EE + n]; fb = FB2; ksteps = 5; }
    int kk = kl >> 5, kr = kl & 31;
    int nt = n >> 4,  m  = n & 15;
    int lane = m + (((kr >> 3) & 1) << 4);
    int elem = ((kr >> 4) << 3) + (kr & 7);
    int frag = fb + nt * ksteps + kk;
    wsb[(size_t)frag * 512 + lane * 16 + elem] = f2bfu(val);
}

// ---------------------------------------------------------------------------
// One GEMM slice: wave computes 4 of the 32 N-tiles, K = KSTEPS*32.
// A fragments hoisted into registers (loaded once per gemm from swizzled LDS).
// ---------------------------------------------------------------------------
template <int KSTEPS>
__device__ __forceinline__ void gemm_part(const unsigned short* sA, int fragA0,
                                          const unsigned short* wsb, int fragB0,
                                          const float* bias, int gemm,
                                          float* sXhn, int wave, int lane,
                                          int m, int half)
{
    const v16bf* aLds = (const v16bf*)sA;        // frag f: elements [f*32 + lane]
    v16bf afrag[KSTEPS];
    #pragma unroll
    for (int k = 0; k < KSTEPS; ++k)
        afrag[k] = aLds[(fragA0 + k) * 32 + lane];    // 2x ds_load_b128

    const v16bf* bGl = (const v16bf*)wsb;
    for (int it = 0; it < 4; ++it) {             // uniform: 4 tiles per wave
        const int nt = wave * 4 + it;
        const v16bf* bp = bGl + ((size_t)(fragB0 + nt * KSTEPS)) * 32 + lane;
        v8f c = {};
        #pragma unroll
        for (int k = 0; k < KSTEPS; ++k) {
            v16bf b = bp[(size_t)k * 32];        // 2x global_load_b128, coalesced
            c = __builtin_amdgcn_wmma_f32_16x16x32_bf16(
                    false, afrag[k], false, b, (short)0, c, false, false);
        }
        const int ncolW = nt * 16 + m;
        const float bv = bias[ncolW];
        const int colOut = gemm * EE + ncolW;
        #pragma unroll
        for (int r = 0; r < 8; ++r)              // C: VGPR r -> row r / r+8
            sXhn[(r + (half ? 8 : 0)) * C3_ + colOut] = c[r] + bv;
    }
}

// ---------------------------------------------------------------------------
// Kernel 2: fused GEMMs + shuffle + depthwise conv + softmax + output.
// ---------------------------------------------------------------------------
extern "C" __global__ __launch_bounds__(THREADS)
void eca_fused_kernel(const float* __restrict__ x,
                      const float* __restrict__ human,
                      const float* __restrict__ nature,
                      const float* __restrict__ bx,
                      const float* __restrict__ bh,
                      const float* __restrict__ bn,
                      const float* __restrict__ conv_w,
                      const float* __restrict__ conv_b,
                      const int*   __restrict__ perm,
                      const unsigned short* __restrict__ wsb,
                      float* __restrict__ out)
{
    extern __shared__ __attribute__((aligned(32))) char smem[];
    unsigned short* sA    = (unsigned short*)(smem + OFF_A);
    int*            sPerm = (int*)  (smem + OFF_PERM);
    int*            sInv  = (int*)  (smem + OFF_INV);
    float*          sXhn  = (float*)(smem + OFF_XHN);   // [16][1536]
    float*          sS    = (float*)(smem + OFF_S);     // [16][512]

    const int tid  = threadIdx.x;
    const int row0 = blockIdx.x * ROWS;

    // ---- Phase 0: perm tables + A tiles converted & pre-swizzled into LDS ---
    // A-fragment layout (16x32 bf16): for K-in-step kr, row r:
    //   half = ((kr&15) >= 8), lane = r + 16*half, elem = 8*(kr>>4) + (kr&7)
    for (int j = tid; j < C3_; j += THREADS) {
        int p = perm[j];
        sPerm[j] = p;
        sInv[p]  = j;
    }
    #pragma unroll 1
    for (int idx = tid; idx < ROWS * XS_; idx += THREADS) {
        int r = idx >> 7, kl = idx & 127;
        int kk = kl >> 5, kr = kl & 31;
        int q = kr & 15, half = q >> 3;
        int lane = r + (half << 4);
        int elem = ((kr >> 4) << 3) + (q & 7);
        sA[(FA0 + kk) * 512 + lane * 16 + elem] = f2bfu(x[(size_t)(row0 + r) * XS_ + kl]);
    }
    #pragma unroll 1
    for (int idx = tid; idx < ROWS * HST; idx += THREADS) {
        int r = idx / HST, kl = idx - r * HST;
        int kk = kl >> 5, kr = kl & 31;
        int q = kr & 15, half = q >> 3;
        int lane = r + (half << 4);
        int elem = ((kr >> 4) << 3) + (q & 7);
        sA[(FA1 + kk) * 512 + lane * 16 + elem] = f2bfu(human[(size_t)(row0 + r) * HST + kl]);
    }
    #pragma unroll 1
    for (int idx = tid; idx < ROWS * NST; idx += THREADS) {
        int r = idx / NST, kl = idx - r * NST;
        int kk = kl >> 5, kr = kl & 31;
        int q = kr & 15, half = q >> 3;
        int lane = r + (half << 4);
        int elem = ((kr >> 4) << 3) + (q & 7);
        sA[(FA2 + kk) * 512 + lane * 16 + elem] = f2bfu(nature[(size_t)(row0 + r) * NST + kl]);
    }
    __syncthreads();

    // ---- Phase 1: three GEMMs, WMMA bf16, fp32 accumulate --------------------
    const int wave = tid >> 5;
    const int lane = tid & 31;
    const int m    = lane & 15;
    const int half = lane >> 4;

    gemm_part<4>(sA, FA0, wsb, FB0, bx, 0, sXhn, wave, lane, m, half);
    gemm_part<5>(sA, FA1, wsb, FB1, bh, 1, sXhn, wave, lane, m, half);
    gemm_part<5>(sA, FA2, wsb, FB2, bn, 2, sXhn, wave, lane, m, half);
    __syncthreads();

    // ---- Phase 2: shuffle + depthwise 5-tap conv (gather form) --------------
    // xx[row][j] = conv_b[t] + sum_k conv_w[t,k] * xhn[row][ perm[inv[j]-3+k] ]
    for (int idx = tid; idx < ROWS * EE; idx += THREADS) {
        const int row = idx >> 9;
        const int j   = idx & 511;
        const int t   = (row0 + row) & 63;
        const int i   = sInv[j];
        float acc = conv_b[t];
        #pragma unroll
        for (int k = 0; k < 5; ++k) {
            int p = i - 3 + k;
            if (p >= 0 && p < C3_)
                acc += conv_w[t * 5 + k] * sXhn[row * C3_ + sPerm[p]];
        }
        sS[idx] = acc;
    }
    __syncthreads();

    // ---- Phase 3: per-row softmax(512) + out = xe*(softmax + 1) -------------
    #pragma unroll
    for (int rr = 0; rr < 2; ++rr) {
        const int row  = wave * 2 + rr;
        const int grow = row0 + row;
        float mx = -3.402823e38f;
        for (int j = lane; j < EE; j += 32) mx = fmaxf(mx, sS[row * EE + j]);
        #pragma unroll
        for (int off = 16; off > 0; off >>= 1) mx = fmaxf(mx, __shfl_xor(mx, off, 32));
        float sum = 0.f;
        for (int j = lane; j < EE; j += 32) {
            float e = __expf(sS[row * EE + j] - mx);
            sS[row * EE + j] = e;
            sum += e;
        }
        #pragma unroll
        for (int off = 16; off > 0; off >>= 1) sum += __shfl_xor(sum, off, 32);
        const float inv = 1.f / sum;
        for (int j = lane; j < EE; j += 32) {
            float xe = sXhn[row * C3_ + j];
            out[(size_t)grow * EE + j] = xe * (sS[row * EE + j] * inv + 1.f);
        }
    }
}

extern "C" void kernel_launch(void* const* d_in, const int* in_sizes, int n_in,
                              void* d_out, int out_size, void* d_ws, size_t ws_size,
                              hipStream_t stream)
{
    const float* x      = (const float*)d_in[0];
    const float* human  = (const float*)d_in[1];
    const float* nature = (const float*)d_in[2];
    const float* Wx     = (const float*)d_in[3];
    const float* bx     = (const float*)d_in[4];
    const float* Wh     = (const float*)d_in[5];
    const float* bh     = (const float*)d_in[6];
    const float* Wn     = (const float*)d_in[7];
    const float* bn     = (const float*)d_in[8];
    const float* conv_w = (const float*)d_in[9];
    const float* conv_b = (const float*)d_in[10];
    const int*   perm   = (const int*)d_in[11];
    float*       out    = (float*)d_out;
    unsigned short* wsb = (unsigned short*)d_ws;

    // Kernel 1: weight repack into bf16 B-fragments (448 KB in d_ws)
    {
        int total = KTOT * EE;
        dim3 grid((total + THREADS - 1) / THREADS), block(THREADS);
        hipLaunchKernelGGL(repack_w_kernel, grid, block, 0, stream, Wx, Wh, Wn, wsb);
    }
    // Kernel 2: fused GEMMs + conv + softmax
    {
        const int n_rows = 256 * 64;                   // B*T = 16384
        dim3 grid(n_rows / ROWS), block(THREADS);      // 1024 x 256
        hipLaunchKernelGGL(eca_fused_kernel, grid, block, SMEM_BYTES, stream,
                           x, human, nature, bx, bh, bn,
                           conv_w, conv_b, perm, wsb, out);
    }
}
